// Trittention_11227044512420
// MI455X (gfx1250) — matrix operand order repaired and checked by reading, not compiled, not run on hardware
//
#include <hip/hip_runtime.h>
#include <hip/hip_bf16.h>

typedef __attribute__((ext_vector_type(16))) __bf16 v16bf;
typedef __attribute__((ext_vector_type(8)))  float  v8f;

#define SQ    256
#define DH    64
#define DM    512
#define NHEAD 8

// =====================================================================
// Shared WMMA GEMM core: C[256 x 64] = A[256 x 512] * B[512 x 64] (+bias)
// A row-major (lda), B row-major (ldb), C row-major (ldc), f32 in/out,
// bf16 fragments staged in LDS in the v_wmma_f32_16x16x32_bf16 layout:
//   A 16x32: lanes0-15 hold K={0..7,16..23}, lanes16-31 K={8..15,24..31}
//   B 32x16: lane = col + 16*((k/8)%2), slot i = k%8 + 8*(k/16) (same k order as A)
// 8 waves: wave w owns row-tiles {2w, 2w+1} x all 4 col-tiles.
// =====================================================================
__device__ __forceinline__ void gemm256x64(
    const float* __restrict__ A, int lda,
    const float* __restrict__ B, int ldb,
    const float* __restrict__ bias,           // 64 entries or nullptr
    float* __restrict__ C, int ldc,
    __bf16* aSt, __bf16* bSt)
{
    const int tid     = threadIdx.x;
    const int wave    = tid >> 5;
    const int lane    = tid & 31;
    const int halfsel = lane >> 4;
    const int col0    = lane & 15;

    v8f acc[2][4];
    #pragma unroll
    for (int i = 0; i < 2; ++i)
        #pragma unroll
        for (int j = 0; j < 4; ++j) acc[i][j] = (v8f){};

    for (int kc = 0; kc < 16; ++kc) {         // K = 512 in chunks of 32
        // stage A chunk: thread stages row s=tid, chunk-local k = 0..31
        {
            const int s = tid, tile = s >> 4, m = s & 15;
            const float* Ar = A + (size_t)s * lda + kc * 32;
            #pragma unroll
            for (int k = 0; k < 32; ++k) {
                const int laneA = m + (((k >> 3) & 1) << 4);
                const int iA    = (k & 7) + ((k >> 4) << 3);
                aSt[(tile << 9) + laneA * 16 + iA] = (__bf16)Ar[k];
            }
        }
        // stage B chunk: 32 k-rows x 64 cols; thread covers (8 k) x 1 col
        {
            const int h = tid & 63, kbase = (tid >> 6) * 8;
            const int tileH = h >> 4, m = h & 15;
            #pragma unroll
            for (int j = 0; j < 8; ++j) {
                const int kk    = kbase + j;
            const int laneB = m + (((kk >> 3) & 1) << 4);
            const int iB    = (kk & 7) + ((kk >> 4) << 3);
                bSt[(tileH << 9) + laneB * 16 + iB] =
                    (__bf16)B[(size_t)(kc * 32 + kk) * ldb + h];
            }
        }
        __syncthreads();

        const v16bf a0 = *(const v16bf*)&aSt[((2 * wave + 0) << 9) + lane * 16];
        const v16bf a1 = *(const v16bf*)&aSt[((2 * wave + 1) << 9) + lane * 16];
        #pragma unroll
        for (int th = 0; th < 4; ++th) {
            const v16bf bf = *(const v16bf*)&bSt[(th << 9) + lane * 16];
            acc[0][th] = __builtin_amdgcn_wmma_f32_16x16x32_bf16(
                             false, a0, false, bf, (short)0, acc[0][th], false, false);
            acc[1][th] = __builtin_amdgcn_wmma_f32_16x16x32_bf16(
                             false, a1, false, bf, (short)0, acc[1][th], false, false);
        }
        __syncthreads();                      // protect staging for next chunk
    }

    // writeback: VGPR g -> row sbase+g, col = col-tile*16 + lane%16
    #pragma unroll
    for (int i = 0; i < 2; ++i) {
        const int sbase = (2 * wave + i) * 16 + (halfsel << 3);
        #pragma unroll
        for (int th = 0; th < 4; ++th) {
            const int h = th * 16 + col0;
            const float bb = bias ? bias[h] : 0.0f;
            #pragma unroll
            for (int g = 0; g < 8; ++g)
                C[(size_t)(sbase + g) * ldc + h] = acc[i][th][g] + bb;
        }
    }
}

// ---------------- kernel 1: the five linear projections (WMMA) ----------------
// blockIdx.x = head n, blockIdx.y selects: 0=k1, 1=k2, 2=q, 3=vA(+b_V12), 4=vB
__global__ __launch_bounds__(256) void proj_wmma(
    const float* __restrict__ x,
    const float* __restrict__ Wk1, const float* __restrict__ Wk2,
    const float* __restrict__ Wq,  const float* __restrict__ Wv12,
    const float* __restrict__ bk1, const float* __restrict__ bk2,
    const float* __restrict__ bq,  const float* __restrict__ bv12,
    float* __restrict__ k1, float* __restrict__ k2, float* __restrict__ qo,
    float* __restrict__ vA, float* __restrict__ vB)
{
    __shared__ __bf16 aSt[16 * 512] __attribute__((aligned(32)));
    __shared__ __bf16 bSt[4 * 512]  __attribute__((aligned(32)));

    const int n = blockIdx.x;
    const float* W; const float* bias; float* out; int Dw, doff;
    switch (blockIdx.y) {
      case 0:  W = Wk1;  bias = bk1;  out = k1; Dw = DM;   doff = 0;  break;
      case 1:  W = Wk2;  bias = bk2;  out = k2; Dw = DM;   doff = 0;  break;
      case 2:  W = Wq;   bias = bq;   out = qo; Dw = DM;   doff = 0;  break;
      case 3:  W = Wv12; bias = bv12; out = vA; Dw = 2*DM; doff = 0;  break;
      default: W = Wv12; bias = nullptr; out = vB; Dw = 2*DM; doff = DM; break;
    }
    const float* B     = W + ((size_t)n * Dw + doff) * DH;  // [512 x 64], ldb=64
    const float* biasp = bias ? bias + n * DH : nullptr;
    float*       C     = out + (size_t)n * SQ * DH;         // [256 x 64], ldc=64
    gemm256x64(x, DM, B, DH, biasp, C, DH, aSt, bSt);
}

// ---------------- kernel 2: per-(head,query) cubic attention ----------------
// One block per (n, q). Streams the 256x256 score tile through WMMA fragments,
// applies mask s<t<q, exp, and accumulates row/col sums of the attention
// matrix (that is all that is needed since v[p,s] = vA[p] + vB[s] + b).
__global__ __launch_bounds__(256) void attn_kernel(
    const float* __restrict__ k1, const float* __restrict__ k2,
    const float* __restrict__ qo,
    const float* __restrict__ vA, const float* __restrict__ vB,
    float* __restrict__ zbuf)
{
    // fragment-ready bf16 staging: [tile(16)][khalf(2)][lane(32)*16 halves]
    __shared__ __bf16 aStage[16 * 2 * 512] __attribute__((aligned(32)));
    __shared__ __bf16 bStage[16 * 2 * 512] __attribute__((aligned(32)));
    __shared__ float  rowsum[SQ];
    __shared__ float  colsum[SQ];
    __shared__ float  red[64];
    __shared__ float  totalSh;

    const int n    = blockIdx.x >> 8;
    const int qidx = blockIdx.x & 255;
    const int tid  = threadIdx.x;

    const float* qrow = qo + ((size_t)n * SQ + qidx) * DH;
    const float* k1n  = k1 + (size_t)n * SQ * DH;
    const float* k2n  = k2 + (size_t)n * SQ * DH;

    colsum[tid] = 0.0f;

    // Stage A = k1 * q_vec / H (1/H score scale folded in) and B = k2, bf16.
    {
        const int s = tid, tile = s >> 4, m = s & 15;
        const float invH = 1.0f / (float)DH;
        #pragma unroll 4
        for (int hh = 0; hh < DH; ++hh) {
            const int khalf = hh >> 5, k = hh & 31;
            const float av = k1n[s * DH + hh] * qrow[hh] * invH;
            const float bv = k2n[s * DH + hh];
            const int laneA = m + (((k >> 3) & 1) << 4);
            const int iA    = (k & 7) + ((k >> 4) << 3);
            aStage[(((tile << 1) + khalf) << 9) + laneA * 16 + iA] = (__bf16)av;
            const int laneB = m + (((k >> 3) & 1) << 4);
            const int iB    = (k & 7) + ((k >> 4) << 3);
            bStage[(((tile << 1) + khalf) << 9) + laneB * 16 + iB] = (__bf16)bv;
        }
    }
    __syncthreads();

    const int wave    = tid >> 5;
    const int lane    = tid & 31;
    const int halfsel = lane >> 4;
    const int col0    = lane & 15;

    float colAcc[16];
    #pragma unroll
    for (int i = 0; i < 16; ++i) colAcc[i] = 0.0f;

    for (int rs = 0; rs < 2; ++rs) {
        const int tr = wave + rs * 8;  // this wave's 16-row strip
        const v16bf a0 = *(const v16bf*)&aStage[((tr * 2 + 0) << 9) + lane * 16];
        const v16bf a1 = *(const v16bf*)&aStage[((tr * 2 + 1) << 9) + lane * 16];
        const int sbase = tr * 16 + (halfsel << 3);

        float rowAcc[8];
        #pragma unroll
        for (int g = 0; g < 8; ++g) rowAcc[g] = 0.0f;

        #pragma unroll
        for (int tc = 0; tc < 16; ++tc) {
            const v16bf b0 = *(const v16bf*)&bStage[((tc * 2 + 0) << 9) + lane * 16];
            const v16bf b1 = *(const v16bf*)&bStage[((tc * 2 + 1) << 9) + lane * 16];
            v8f acc = {};
            acc = __builtin_amdgcn_wmma_f32_16x16x32_bf16(
                      false, a0, false, b0, (short)0, acc, false, false);
            acc = __builtin_amdgcn_wmma_f32_16x16x32_bf16(
                      false, a1, false, b1, (short)0, acc, false, false);

            const int  tcol     = tc * 16 + col0;
            const bool colValid = (tcol < qidx);
            float csum = 0.0f;
            #pragma unroll
            for (int g = 0; g < 8; ++g) {
                const int srow = sbase + g;
                float e = __expf(acc[g]);                       // arg already /H
                e = (colValid && (srow < tcol)) ? e : 0.0f;     // mask: s < t < q
                rowAcc[g] += e;
                csum += e;
            }
            colAcc[tc] += csum;
        }

        // butterfly-reduce rows over the 16 lanes of this half, plain store
        #pragma unroll
        for (int g = 0; g < 8; ++g) {
            float v = rowAcc[g];
            v += __shfl_xor(v, 1, 32);
            v += __shfl_xor(v, 2, 32);
            v += __shfl_xor(v, 4, 32);
            v += __shfl_xor(v, 8, 32);
            rowAcc[g] = v;
        }
        if (col0 == 0) {
            #pragma unroll
            for (int g = 0; g < 8; ++g) rowsum[sbase + g] = rowAcc[g];
        }
    }

    // columns: reduce lane<->lane+16, then 16 ds_add_f32 per wave
    #pragma unroll
    for (int tc = 0; tc < 16; ++tc) {
        const float v = colAcc[tc] + __shfl_xor(colAcc[tc], 16, 32);
        if (halfsel == 0) atomicAdd(&colsum[tc * 16 + col0], v);
    }
    __syncthreads();

    // total = sum of all exp values
    if (tid < 64) {
        float p = 0.0f;
        #pragma unroll
        for (int j = 0; j < 4; ++j) p += rowsum[tid * 4 + j];
        red[tid] = p;
    }
    __syncthreads();
    if (tid == 0) {
        float t = 0.0f;
        for (int j = 0; j < 64; ++j) t += red[j];
        totalSh = t;
    }
    __syncthreads();

    // z[q, n*H+h] = sum_p r[p]*vA[p,h] + sum_s c[s]*vB[s,h]  (b_V12 in vA)
    if (tid < DH) {
        const int h = tid;
        const float tot = totalSh;
        const bool  uni = !(tot > 0.0f);       // fully-masked (q<2): uniform attn
        const float inv = uni ? 0.0f : 1.0f / tot;
        const float* vAn = vA + (size_t)n * SQ * DH + h;
        const float* vBn = vB + (size_t)n * SQ * DH + h;
        float acc = 0.0f;
        for (int p = 0; p < SQ; ++p) {
            const float rn = uni ? (1.0f / 256.0f) : rowsum[p] * inv;
            const float cn = uni ? (1.0f / 256.0f) : colsum[p] * inv;
            acc += rn * vAn[p * DH] + cn * vBn[p * DH];
        }
        // row-major [q][n*H + h] so the output projection is a plain GEMM
        zbuf[(size_t)qidx * (NHEAD * DH) + n * DH + h] = acc;
    }
}

// ---------------- kernel 3: output projection (WMMA) ----------------
// out[256 x 512] = z[256 x 512] @ WO[512 x 512] + bO; one block per 64-col slice
__global__ __launch_bounds__(256) void out_wmma(
    const float* __restrict__ z, const float* __restrict__ WO,
    const float* __restrict__ bO, float* __restrict__ out)
{
    __shared__ __bf16 aSt[16 * 512] __attribute__((aligned(32)));
    __shared__ __bf16 bSt[4 * 512]  __attribute__((aligned(32)));
    const int b = blockIdx.x;                  // 0..7
    gemm256x64(z, NHEAD * DH, WO + b * 64, DM, bO + b * 64,
               out + b * 64, DM, aSt, bSt);
}

extern "C" void kernel_launch(void* const* d_in, const int* in_sizes, int n_in,
                              void* d_out, int out_size, void* d_ws, size_t ws_size,
                              hipStream_t stream) {
    const float* x    = (const float*)d_in[0];
    const float* Wk1  = (const float*)d_in[1];
    const float* Wk2  = (const float*)d_in[2];
    const float* Wq   = (const float*)d_in[3];
    const float* Wv12 = (const float*)d_in[4];
    const float* WO   = (const float*)d_in[5];
    const float* bk1  = (const float*)d_in[6];
    const float* bk2  = (const float*)d_in[7];
    const float* bq   = (const float*)d_in[8];
    const float* bv12 = (const float*)d_in[9];
    const float* bO   = (const float*)d_in[10];
    float* out = (float*)d_out;

    float* ws = (float*)d_ws;
    const size_t PER = (size_t)NHEAD * SQ * DH;   // 131072 floats
    float* k1 = ws;
    float* k2 = ws + PER;
    float* qo = ws + 2 * PER;
    float* vA = ws + 3 * PER;
    float* vB = ws + 4 * PER;
    float* zb = ws + 5 * PER;                     // z as [256][512] row-major

    proj_wmma<<<dim3(NHEAD, 5), 256, 0, stream>>>(x, Wk1, Wk2, Wq, Wv12,
                                                  bk1, bk2, bq, bv12,
                                                  k1, k2, qo, vA, vB);
    attn_kernel<<<NHEAD * SQ, 256, 0, stream>>>(k1, k2, qo, vA, vB, zb);
    out_wmma<<<DM / 64, 256, 0, stream>>>(zb, WO, bO, out);

    (void)in_sizes; (void)n_in; (void)out_size; (void)ws_size;
}
